// SpatialAttentionBlock_12086037971182
// MI455X (gfx1250) — matrix operand order repaired and checked by reading, not compiled, not run on hardware
//
#include <hip/hip_runtime.h>
#include <hip/hip_bf16.h>

// SpatialAttentionBlock for MI455X (gfx1250, wave32, WMMA).
// Pipeline: mask_prep -> conv_qk (3x3 convs, fp32 VALU, LDS-staged rows)
//        -> flash_attn (WMMA f16, online softmax, 32-key tiles staged in LDS
//                       via GLOBAL_LOAD_ASYNC_TO_LDS when available)
//        -> fuse_conv (1x1 conv as WMMA GEMM, weights staged in LDS).

typedef __attribute__((ext_vector_type(16))) _Float16 v16h;
typedef __attribute__((ext_vector_type(8)))  float    v8f;
typedef __attribute__((ext_vector_type(4)))  int      i32x4;

#define BB   8
#define CINX 64
#define CC   96      // C32 = 96 attention channels (= 3 x 32 WMMA k-steps)
#define DWN  64      // downsampled H=W
#define N64  4096    // 64*64 tokens
#define HH   256
#define WWD  256
#define HW   65536
#define CMID 64
#define KTOT 160     // 64 (x) + 96 (k_up) channels into final 1x1 conv

// Async global->LDS (CDNA5): guarded, with synchronous fallback.
#if defined(__has_builtin)
#if __has_builtin(__builtin_amdgcn_global_load_async_to_lds_b128) && \
    __has_builtin(__builtin_amdgcn_s_wait_asynccnt)
#define HAVE_ASYNC_LDS 1
#endif
#endif
#ifndef HAVE_ASYNC_LDS
#define HAVE_ASYNC_LDS 0
#endif

#if HAVE_ASYNC_LDS
// Builtin signature (from hipcc diagnostic): (int4 AS1*, int4 AS3*, imm, imm)
#define GAS_I4(p) ((__attribute__((address_space(1))) i32x4*)(p))
#define LAS_I4(p) ((__attribute__((address_space(3))) i32x4*)(p))
#endif

// ---- WMMA fragment index maps (from CDNA5 ISA 7.12.2, wave32) -------------
// A (16x32 f16): lane row = lane&15; K per half:
__device__ __forceinline__ int amap_k(int lane, int e) {
  int v = e >> 1, p = e & 1;
  int kb = (lane < 16) ? 0 : 8;
  return (v < 4) ? (kb + 2 * v + p) : (kb + 16 + 2 * (v - 4) + p);
}
// B (32x16 f16): col = lane&15; K per half same as A:
__device__ __forceinline__ int bmap_k(int lane, int e) {
  int v = e >> 1, p = e & 1;
  int kb = (lane < 16) ? 0 : 8;
  return (v < 4) ? (kb + 2 * v + p) : (kb + 16 + 2 * (v - 4) + p);
}
__device__ __forceinline__ v8f wmma_f16(v16h a, v16h b, v8f c) {
  return __builtin_amdgcn_wmma_f32_16x16x32_f16(false, a, false, b, (short)0, c,
                                                false, false);
}
__device__ __forceinline__ v8f vzero8() {
  v8f z;
#pragma unroll
  for (int i = 0; i < 8; ++i) z[i] = 0.f;
  return z;
}

// ---------------------------------------------------------------------------
// Kernel 1: downsampled masks. qm = shadow queries, km = non-shadow keys.
__global__ void mask_prep_kernel(const float* __restrict__ mask,
                                 const float* __restrict__ mask_d,
                                 float* __restrict__ qm,
                                 float* __restrict__ km) {
  int idx = blockIdx.x * 256 + threadIdx.x;
  if (idx >= BB * N64) return;
  int b = idx >> 12, n = idx & 4095;
  int i = n >> 6, j = n & 63;
  size_t src = (size_t)b * HW + (size_t)(4 * i) * WWD + 4 * j;  // nearest: *4
  float m = mask[src], md = mask_d[src];
  qm[idx] = (m == 1.0f) ? 1.f : 0.f;
  km[idx] = ((md - m) == 1.0f) ? 1.f : 0.f;
}

// ---------------------------------------------------------------------------
// Kernel 2: both 3x3 convs (reflect pad) on the downsampled image.
// One block per (batch, output row). Stages 3 input rows (all 64 cin) in LDS.
// Thread: j = tid&63 (output col), sub = tid>>6; computes co = sub+4k, k<24,
// for both weight sets -> weight addresses are wave-uniform.
__global__ __launch_bounds__(256) void conv_qk_kernel(
    const float* __restrict__ x, const float* __restrict__ w_s,
    const float* __restrict__ b_s, const float* __restrict__ w_ns,
    const float* __restrict__ b_ns, _Float16* __restrict__ Qh,
    _Float16* __restrict__ Kh) {
  __shared__ float xls[3 * CINX * DWN];  // [dy][cin][j]
  int b = blockIdx.x >> 6;
  int i = blockIdx.x & 63;
  int tid = threadIdx.x;
  for (int t = tid; t < 3 * CINX * DWN; t += 256) {
    int dy = t / (CINX * DWN);
    int rem = t - dy * (CINX * DWN);
    int cin = rem >> 6;
    int j = rem & 63;
    int ir = i + dy - 1;                       // reflect pad 1
    ir = ir < 0 ? -ir : (ir > 63 ? 126 - ir : ir);
    xls[t] = x[(((size_t)b * CINX + cin) * HH + (size_t)(4 * ir)) * WWD + 4 * j];
  }
  __syncthreads();

  int j = tid & 63;
  int sub = tid >> 6;  // 0..3
  float accS[24], accN[24];
#pragma unroll
  for (int k = 0; k < 24; ++k) {
    accS[k] = b_s[sub + 4 * k];
    accN[k] = b_ns[sub + 4 * k];
  }
  for (int cin = 0; cin < CINX; ++cin) {
#pragma unroll
    for (int dy = 0; dy < 3; ++dy) {
      const float* xrow = &xls[(dy * CINX + cin) * DWN];
#pragma unroll
      for (int dx = 0; dx < 3; ++dx) {
        int jj = j + dx - 1;
        jj = jj < 0 ? -jj : (jj > 63 ? 126 - jj : jj);
        float xv = xrow[jj];
        int widx = cin * 9 + dy * 3 + dx;
#pragma unroll
        for (int k = 0; k < 24; ++k) {
          int co = sub + 4 * k;
          accS[k] = fmaf(xv, w_s[(size_t)co * (CINX * 9) + widx], accS[k]);
          accN[k] = fmaf(xv, w_ns[(size_t)co * (CINX * 9) + widx], accN[k]);
        }
      }
    }
  }
  int n = i * 64 + j;
  size_t base = ((size_t)b * N64 + n) * CC;
#pragma unroll
  for (int k = 0; k < 24; ++k) {
    int co = sub + 4 * k;
    float qs = accS[k]; qs = qs > 0.f ? qs : 0.2f * qs;  // LeakyReLU(0.2)
    float kn = accN[k]; kn = kn > 0.f ? kn : 0.2f * kn;
    Qh[base + co] = (_Float16)qs;
    Kh[base + co] = (_Float16)kn;
  }
}

// ---------------------------------------------------------------------------
// Kernel 3: flash attention. Block = 8 waves; wave owns 16 query rows.
// Loop over 128 key tiles of 32 (LDS-staged, async on CDNA5). Per tile:
// S = Q K^T via 2x3 WMMA, masked online softmax, O += P V via 6 WMMA.
__global__ __launch_bounds__(256) void flash_attn_kernel(
    const _Float16* __restrict__ Qh, const _Float16* __restrict__ Kh,
    const float* __restrict__ qm, const float* __restrict__ km,
    _Float16* __restrict__ Knew) {
  __shared__ __align__(16) _Float16 kT[32 * CC];     // key tile [key][c], 6144B
  __shared__ __align__(16) float kmT[32];            // key mask tile, 128B
  __shared__ _Float16 pS[8][16 * 32];                // per-wave P scratch
  int b = blockIdx.x >> 5;
  int qblk = blockIdx.x & 31;
  int wave = threadIdx.x >> 5;
  int lane = threadIdx.x & 31;
  int qbase = qblk * 128 + wave * 16;
  int colHalf = lane & 15;
  int half8 = (lane < 16) ? 0 : 8;

  // Q fragments for this wave's 16 rows (channels 0..95 = 3 k-steps).
  v16h qf[3];
  {
    size_t qoff = ((size_t)b * N64 + (qbase + colHalf)) * CC;
#pragma unroll
    for (int ks = 0; ks < 3; ++ks)
#pragma unroll
      for (int e = 0; e < 16; ++e)
        qf[ks][e] = Qh[qoff + ks * 32 + amap_k(lane, e)];
  }

  v8f O[6];
#pragma unroll
  for (int c = 0; c < 6; ++c) O[c] = vzero8();
  float mrow[8], lrow[8];
#pragma unroll
  for (int v = 0; v < 8; ++v) { mrow[v] = -1e30f; lrow[v] = 0.f; }
  const float scale = 0.1020620726f;  // 1/sqrt(96)

  for (int kt = 0; kt < 128; ++kt) {
    int kb0 = kt * 32;
    __syncthreads();  // previous tile fully consumed before overwrite
#if HAVE_ASYNC_LDS
    {
      // 6144B key tile = 384 x b128 chunks, direct global->LDS (ASYNCcnt).
      const char* src = (const char*)&Kh[((size_t)b * N64 + kb0) * CC];
      for (int t = threadIdx.x; t < 384; t += 256)
        __builtin_amdgcn_global_load_async_to_lds_b128(
            GAS_I4(src + (size_t)t * 16), LAS_I4((char*)kT + t * 16), 0, 0);
      if (threadIdx.x < 8) {
        const char* ms = (const char*)&km[b * N64 + kb0];
        __builtin_amdgcn_global_load_async_to_lds_b128(
            GAS_I4(ms + threadIdx.x * 16),
            LAS_I4((char*)kmT + threadIdx.x * 16), 0, 0);
      }
      __builtin_amdgcn_s_wait_asynccnt(0);  // this wave's chunks landed
    }
#else
    // [key][c] block is contiguous in Kh -> fully coalesced.
    for (int t = threadIdx.x; t < 32 * CC; t += 256)
      kT[t] = Kh[((size_t)b * N64 + kb0) * CC + t];
    if (threadIdx.x < 32) kmT[threadIdx.x] = km[b * N64 + kb0 + threadIdx.x];
#endif
    __syncthreads();
    if (kt + 1 < 128)  // global_prefetch_b8 for next key tile
      __builtin_prefetch(
          (const char*)&Kh[((size_t)b * N64 + kb0 + 32) * CC] + threadIdx.x * 48,
          0, 1);

    // ---- S = Q K^T : two 16x16 D-tiles over 32 keys, 3 k-steps each ----
    v8f s0 = vzero8(), s1 = vzero8();
#pragma unroll
    for (int ks = 0; ks < 3; ++ks) {
      v16h b0, b1;
#pragma unroll
      for (int e = 0; e < 16; ++e) {
        int kch = ks * 32 + bmap_k(lane, e);
        b0[e] = kT[colHalf * CC + kch];         // keys 0..15
        b1[e] = kT[(16 + colHalf) * CC + kch];  // keys 16..31
      }
      s0 = wmma_f16(qf[ks], b0, s0);
      s1 = wmma_f16(qf[ks], b1, s1);
    }

    // ---- masked online softmax (rows live in VGPR dim of D-layout) ----
    float km0 = kmT[colHalf], km1 = kmT[16 + colHalf];
    float alpha[8];
#pragma unroll
    for (int v = 0; v < 8; ++v) {
      float a = km0 > 0.5f ? s0[v] * scale : -1e30f;
      float c = km1 > 0.5f ? s1[v] * scale : -1e30f;
      float t = fmaxf(a, c);  // row-max across the 16 lanes of this half
      t = fmaxf(t, __shfl_xor(t, 1, 32));
      t = fmaxf(t, __shfl_xor(t, 2, 32));
      t = fmaxf(t, __shfl_xor(t, 4, 32));
      t = fmaxf(t, __shfl_xor(t, 8, 32));
      float mnew = fmaxf(mrow[v], t);
      alpha[v] = __expf(mrow[v] - mnew);
      mrow[v] = mnew;
    }
#pragma unroll
    for (int v = 0; v < 8; ++v) {
      float p0 = km0 > 0.5f ? __expf(s0[v] * scale - mrow[v]) : 0.f;
      float p1 = km1 > 0.5f ? __expf(s1[v] * scale - mrow[v]) : 0.f;
      int row = v + half8;
      pS[wave][row * 32 + colHalf] = (_Float16)p0;       // D->A transpose
      pS[wave][row * 32 + 16 + colHalf] = (_Float16)p1;  // via LDS
      float ps = p0 + p1;
      ps += __shfl_xor(ps, 1, 32);
      ps += __shfl_xor(ps, 2, 32);
      ps += __shfl_xor(ps, 4, 32);
      ps += __shfl_xor(ps, 8, 32);
      lrow[v] = lrow[v] * alpha[v] + ps;
    }
#pragma unroll
    for (int c = 0; c < 6; ++c)
#pragma unroll
      for (int v = 0; v < 8; ++v) O[c][v] *= alpha[v];

    // ---- O += P V : A = P (16x32), B = V tile (32 x 96) ----
    v16h pa;
#pragma unroll
    for (int e = 0; e < 16; ++e)
      pa[e] = pS[wave][colHalf * 32 + amap_k(lane, e)];  // same-wave LDS: in-order
#pragma unroll
    for (int ct = 0; ct < 6; ++ct) {
      v16h vb;
#pragma unroll
      for (int e = 0; e < 16; ++e)
        vb[e] = kT[bmap_k(lane, e) * CC + ct * 16 + colHalf];
      O[ct] = wmma_f16(pa, vb, O[ct]);
    }
  }

  // ---- epilogue: k_new = qmask ? O/l : k_ns ----
#pragma unroll
  for (int ct = 0; ct < 6; ++ct)
#pragma unroll
    for (int v = 0; v < 8; ++v) {
      int n = qbase + v + half8;
      int c = ct * 16 + colHalf;
      size_t idx = ((size_t)b * N64 + n) * CC + c;
      float val = (qm[b * N64 + n] > 0.5f) ? (O[ct][v] / lrow[v])
                                           : (float)Kh[idx];
      Knew[idx] = (_Float16)val;
    }
}

// ---------------------------------------------------------------------------
// Kernel 4: final 1x1 conv over concat([x, nearest_up(k_new)]) as WMMA GEMM.
// Block = 8 waves = 128 pixels; weights (160x64) staged once in LDS.
// Per wave: 5 k-steps x 4 n-tiles = 20 WMMAs for 16 pixels x 64 out-channels.
__global__ __launch_bounds__(256) void fuse_conv_kernel(
    const float* __restrict__ x, const _Float16* __restrict__ Knew,
    const float* __restrict__ w_f, const float* __restrict__ b_f,
    float* __restrict__ out) {
  __shared__ _Float16 wl[KTOT * CMID];  // [ci][co]
  int b = blockIdx.x >> 9;
  int pblk = blockIdx.x & 511;
  for (int t = threadIdx.x; t < KTOT * CMID; t += 256) {
    int ci = t >> 6, co = t & 63;
    wl[t] = (_Float16)w_f[(size_t)co * KTOT + ci];
  }
  __syncthreads();

  int wave = threadIdx.x >> 5, lane = threadIdx.x & 31;
  int colHalf = lane & 15;
  int pbase = pblk * 128 + wave * 16;
  int pix = pbase + colHalf;  // A-matrix row pixel for this lane
  int py = pix >> 8, px = pix & 255;
  int n64 = (py >> 2) * 64 + (px >> 2);  // nearest upsample source token
  size_t kbase = ((size_t)b * N64 + n64) * CC;
  size_t xbase = (size_t)b * CINX * HW + pix;

  v8f acc[4];
#pragma unroll
  for (int nt = 0; nt < 4; ++nt) acc[nt] = vzero8();

#pragma unroll
  for (int ks = 0; ks < 5; ++ks) {
    v16h af;
#pragma unroll
    for (int e = 0; e < 16; ++e) {
      int ci = ks * 32 + amap_k(lane, e);
      af[e] = (ci < CINX) ? (_Float16)x[xbase + (size_t)ci * HW]
                          : Knew[kbase + (ci - CINX)];
    }
#pragma unroll
    for (int nt = 0; nt < 4; ++nt) {
      v16h bf;
#pragma unroll
      for (int e = 0; e < 16; ++e)
        bf[e] = wl[(ks * 32 + bmap_k(lane, e)) * CMID + nt * 16 + colHalf];
      acc[nt] = wmma_f16(af, bf, acc[nt]);
    }
  }

  int half8 = (lane < 16) ? 0 : 8;
#pragma unroll
  for (int nt = 0; nt < 4; ++nt) {
    int co = nt * 16 + colHalf;
    float bias = b_f[co];
#pragma unroll
    for (int v = 0; v < 8; ++v) {
      int p = pbase + v + half8;
      float val = acc[nt][v] + bias;
      val = val > 0.f ? val : 0.2f * val;  // LeakyReLU(0.2)
      out[((size_t)b * CMID + co) * HW + p] = val;
    }
  }
}

// ---------------------------------------------------------------------------
extern "C" void kernel_launch(void* const* d_in, const int* in_sizes, int n_in,
                              void* d_out, int out_size, void* d_ws,
                              size_t ws_size, hipStream_t stream) {
  (void)in_sizes; (void)n_in; (void)out_size; (void)ws_size;
  const float* x      = (const float*)d_in[0];
  const float* mask   = (const float*)d_in[1];
  const float* mask_d = (const float*)d_in[2];
  const float* w_s    = (const float*)d_in[3];
  const float* b_s    = (const float*)d_in[4];
  const float* w_ns   = (const float*)d_in[5];
  const float* b_ns   = (const float*)d_in[6];
  const float* w_f    = (const float*)d_in[7];
  const float* b_f    = (const float*)d_in[8];
  float* out = (float*)d_out;

  // Workspace layout (~19.1 MB, L2-resident on MI455X):
  char* ws = (char*)d_ws;
  size_t szh = (size_t)BB * N64 * CC * sizeof(_Float16);  // 6.29 MB each
  _Float16* Qh   = (_Float16*)ws;
  _Float16* Kh   = (_Float16*)(ws + szh);
  _Float16* Knew = (_Float16*)(ws + 2 * szh);
  float* qm = (float*)(ws + 3 * szh);
  float* km = (float*)(ws + 3 * szh + (size_t)BB * N64 * sizeof(float));

  mask_prep_kernel<<<(BB * N64 + 255) / 256, 256, 0, stream>>>(mask, mask_d,
                                                               qm, km);
  conv_qk_kernel<<<BB * DWN, 256, 0, stream>>>(x, w_s, b_s, w_ns, b_ns, Qh, Kh);
  flash_attn_kernel<<<BB * 32, 256, 0, stream>>>(Qh, Kh, qm, km, Knew);
  fuse_conv_kernel<<<BB * 512, 256, 0, stream>>>(x, Knew, w_f, b_f, out);
}
